// CRF_8907762172010
// MI455X (gfx1250) — compile-verified, matrix operand order checked by reading, not run on hardware
//
#include <hip/hip_runtime.h>

// ---------------------------------------------------------------------------
// CRF forward (log-semiring scan) on MI455X / gfx1250.
//   fv_{t+1}[n] = m + log( sum_p exp(T[n,p]) * exp(fv_t[p]-m) ) + h[t][n]
// exp(T) precomputed in f16, pre-swizzled into WMMA-B tile layout, resident in
// LDS (256KB slice per WGP). Inner product = v_wmma_f32_16x16x32_f16.
// 32 persistent workgroups, sense-reversing global barrier per step.
// ---------------------------------------------------------------------------

#define K_TAGS   2048
#define SEQ_LEN  8192
#define NEG_VAL  (-10000.0f)
#define NUM_WG   32
#define THREADS  1024     // 32 waves of 32
#define NT_PER_WG 64      // next tags per workgroup (4 tiles of 16)
#define NTILES    4
#define KCHUNKS   64      // 2048 / 32

// LDS layout (dynamic shared)
#define EXP_BYTES   (NTILES * KCHUNKS * 1024)   // 262144: [tile][kc][lane(32)][32B]
#define P_OFF       EXP_BYTES                   // 2048 f16 = 4096 B
#define PARTIAL_OFF (P_OFF + K_TAGS * 2)        // 64 f32
#define WMAX_OFF    (PARTIAL_OFF + 64 * 4)      // 32 f32
#define BCAST_OFF   (WMAX_OFF + 32 * 4)         // 1 f32
#define SMEM_BYTES  (BCAST_OFF + 64)

typedef __attribute__((ext_vector_type(16))) _Float16 v16h;
typedef __attribute__((ext_vector_type(8)))  float    v8f;

union ABReg { uint4 q[2]; v16h h; };

__device__ __forceinline__ unsigned pack2h(float a, float b) {
  union { _Float16 h[2]; unsigned u; } p;
  p.h[0] = (_Float16)a; p.h[1] = (_Float16)b;
  return p.u;
}

__device__ __forceinline__ float waveMax(float v) {
  #pragma unroll
  for (int m = 1; m <= 16; m <<= 1) v = fmaxf(v, __shfl_xor(v, m, 32));
  return v;
}
__device__ __forceinline__ float waveSum(float v) {
  #pragma unroll
  for (int m = 1; m <= 16; m <<= 1) v += __shfl_xor(v, m, 32);
  return v;
}

__device__ __forceinline__ float blockMax(float v, float* wred, float* bcast) {
  int wave = threadIdx.x >> 5, lane = threadIdx.x & 31;
  v = waveMax(v);
  if (lane == 0) wred[wave] = v;
  __syncthreads();
  if (wave == 0) {
    float x = wred[lane];           // 32 waves exactly
    x = waveMax(x);
    if (lane == 0) *bcast = x;
  }
  __syncthreads();
  return *bcast;
}

__device__ __forceinline__ float blockSum(float v, float* wred, float* bcast) {
  int wave = threadIdx.x >> 5, lane = threadIdx.x & 31;
  v = waveSum(v);
  if (lane == 0) wred[wave] = v;
  __syncthreads();
  if (wave == 0) {
    float x = wred[lane];
    x = waveSum(x);
    if (lane == 0) *bcast = x;
  }
  __syncthreads();
  return *bcast;
}

// Sense-reversing global barrier across NUM_WG workgroups. Target sense
// alternates deterministically with `round`, so no per-WG sense storage.
__device__ __forceinline__ void gridBarrier(unsigned* cnt, unsigned* sense,
                                            unsigned round) {
  __threadfence();
  __syncthreads();
  if (threadIdx.x == 0) {
    unsigned target = (round & 1u) ? 0u : 1u;
    unsigned prev = __hip_atomic_fetch_add(cnt, 1u, __ATOMIC_ACQ_REL,
                                           __HIP_MEMORY_SCOPE_AGENT);
    if (prev == NUM_WG - 1) {
      __hip_atomic_store(cnt, 0u, __ATOMIC_RELAXED, __HIP_MEMORY_SCOPE_AGENT);
      __hip_atomic_store(sense, target, __ATOMIC_RELEASE,
                         __HIP_MEMORY_SCOPE_AGENT);
    } else {
      while (__hip_atomic_load(sense, __ATOMIC_ACQUIRE,
                               __HIP_MEMORY_SCOPE_AGENT) != target)
        __builtin_amdgcn_s_sleep(1);
    }
  }
  __syncthreads();
  __threadfence();
}

__global__ void crf_init(float* fvbuf, unsigned* bar) {
  int tid = threadIdx.x;
  if (tid < 2) bar[tid] = 0u;
  fvbuf[tid]        = (tid == 0) ? 0.0f : NEG_VAL;  // START=0 gets 0
  fvbuf[tid + 1024] = NEG_VAL;
}

__global__ void __launch_bounds__(THREADS, 1)
crf_scan(const float* __restrict__ h, const float* __restrict__ T,
         float* __restrict__ out, float* fvbuf, unsigned* bar) {
  extern __shared__ char smem[];
  _Float16* pLds    = (_Float16*)(smem + P_OFF);
  float*    partial = (float*)(smem + PARTIAL_OFF);
  float*    wred    = (float*)(smem + WMAX_OFF);
  float*    bcast   = (float*)(smem + BCAST_OFF);

  const int tid  = threadIdx.x;
  const int wg   = blockIdx.x;
  const int wave = tid >> 5;
  const int lane = tid & 31;

  // ------------------------------------------------------------------
  // Fill LDS with exp(T) slice in WMMA-B tile layout (one time).
  // B (K=32 prev x N=16 next) per-lane halves, mirroring the documented
  // 16-bit A layout: lane n<16 -> col n, K {0..7,16..23};
  //                  lane n+16 -> col n, K {8..15,24..31}.
  // Each run = 8 consecutive prevs (16B) for one (tile,kc,lane,half).
  // ------------------------------------------------------------------
  #pragma unroll 4
  for (int i = 0; i < 16; ++i) {
    int r    = tid + (i << 10);        // 16384 runs total
    int h2   = r & 1;
    int ln   = (r >> 1) & 31;
    int kc   = (r >> 6) & 63;
    int tb   = r >> 12;                // 0..3
    int next = wg * NT_PER_WG + tb * 16 + (ln & 15);
    int pbase = (kc << 5) + (h2 << 4) + ((ln >> 4) << 3);
    const float4* src = (const float4*)(T + (size_t)next * K_TAGS + pbase);
    float4 x0 = src[0], x1 = src[1];
    uint4 d;
    d.x = pack2h(__expf(x0.x), __expf(x0.y));
    d.y = pack2h(__expf(x0.z), __expf(x0.w));
    d.z = pack2h(__expf(x1.x), __expf(x1.y));
    d.w = pack2h(__expf(x1.z), __expf(x1.w));
    *(uint4*)(smem + (((tb << 6) + kc) << 10) + ln * 32 + h2 * 16) = d;
  }
  __syncthreads();

  const int tb    = wave & 3;     // which next-tile this wave owns
  const int kslot = wave >> 2;    // 8 waves share a tile; 8 kc each
  // A operand source offsets in pLds (bytes): replicated rows, per-lane-half.
  const int aSel  = (lane >> 4) << 4;   // 0 or 16 bytes within 64B chunk

  for (int t = 0; t < SEQ_LEN; ++t) {
    const float* fv_in  = fvbuf + ((t & 1) ? K_TAGS : 0);
    float*       fv_out = fvbuf + ((t & 1) ? 0 : K_TAGS);

    // ---- m = max(fv); p = exp(fv - m) in f16 -> LDS ----
    float v0 = fv_in[tid];
    float v1 = fv_in[tid + 1024];
    float m  = blockMax(fmaxf(v0, v1), wred, bcast);
    pLds[tid]        = (_Float16)__expf(v0 - m);
    pLds[tid + 1024] = (_Float16)__expf(v1 - m);
    if (tid < 64) partial[tid] = 0.0f;
    __syncthreads();

    // ---- GEMV via WMMA: acc over this wave's 8 prev-chunks ----
    v8f acc = {};
    const char* pB = smem + P_OFF;
    #pragma unroll
    for (int i = 0; i < 8; ++i) {
      int kc = (kslot << 3) + i;
      ABReg a, b;
      int ao = (kc << 6) + aSel;                    // p chunk, this half
      a.q[0] = *(const uint4*)(pB + ao);
      a.q[1] = *(const uint4*)(pB + ao + 32);
      const char* eB = smem + (((tb << 6) + kc) << 10) + lane * 32;
      b.q[0] = *(const uint4*)(eB);
      b.q[1] = *(const uint4*)(eB + 16);
      acc = __builtin_amdgcn_wmma_f32_16x16x32_f16(
          false, a.h, false, b.h, (short)0, acc, false, false);
    }
    // D: all M-rows identical (A rows replicated); lane<16 holds col n=lane.
    if (lane < 16) atomicAdd(&partial[(tb << 4) + lane], acc[0]);
    __syncthreads();

    // ---- fv_out = m + log(S) + emit ----
    if (tid < 64) {
      int nx = wg * NT_PER_WG + tid;
      float s = partial[tid];
      fv_out[nx] = m + __logf(s) + h[(size_t)t * K_TAGS + nx];
      if (t + 1 < SEQ_LEN)
        __builtin_prefetch(h + (size_t)(t + 1) * K_TAGS + nx, 0, 0);
    }
    gridBarrier(bar, bar + 1, (unsigned)t);
  }

  // ---- terminal: out = LSE(fv_final + T[END,:]) ; fv_final in buf 0 ----
  if (wg == 0) {
    const float* fvf = fvbuf;                        // 8192 even -> slot 0
    const float* tEnd = T + (size_t)1 * K_TAGS;      // END_IDX = 1
    float a0 = fvf[tid]        + tEnd[tid];
    float a1 = fvf[tid + 1024] + tEnd[tid + 1024];
    float mm = blockMax(fmaxf(a0, a1), wred, bcast);
    float s  = __expf(a0 - mm) + __expf(a1 - mm);
    float tot = blockSum(s, wred, bcast);
    if (tid == 0) out[0] = mm + __logf(tot);
  }
}

extern "C" void kernel_launch(void* const* d_in, const int* in_sizes, int n_in,
                              void* d_out, int out_size, void* d_ws,
                              size_t ws_size, hipStream_t stream) {
  (void)in_sizes; (void)n_in; (void)out_size; (void)ws_size;
  const float* h = (const float*)d_in[0];   // [8192, 2048] f32
  const float* T = (const float*)d_in[1];   // [2048, 2048] f32
  float* out = (float*)d_out;               // scalar f32

  unsigned* bar = (unsigned*)d_ws;                       // 2 words
  float* fvbuf  = (float*)((char*)d_ws + 256);           // 2 x 2048 f32

  crf_init<<<1, THREADS, 0, stream>>>(fvbuf, bar);
  crf_scan<<<NUM_WG, THREADS, SMEM_BYTES, stream>>>(h, T, out, fvbuf, bar);
}
